// processing_layers_89885075570997
// MI455X (gfx1250) — matrix operand order repaired and checked by reading, not compiled, not run on hardware
//
#include <hip/hip_runtime.h>
#include <math.h>

// ---- model dims (fixed by the reference) ----
constexpr int kB    = 2;
constexpr int kNper = 4096;
constexpr int kNtok = kB * kNper;   // 8192
constexpr int kDin  = 128;
constexpr int kD    = 256;
constexpr int kFF   = 1024;
constexpr int kH    = 8;
constexpr int kHD   = 32;
constexpr int kNH   = 7;
constexpr int kS    = 16;
constexpr int kPED  = 32;
constexpr int kL    = 2;

typedef __attribute__((ext_vector_type(16))) __bf16 v16bf;
typedef __attribute__((ext_vector_type(8)))  float  v8f;

union FragB16 { v16bf v; unsigned int u[8]; };
union FragF32 { v8f  v; float        f[8]; };

__device__ __forceinline__ unsigned short f2bf(float x) {
  unsigned int u = __float_as_uint(x);
  u += 0x7fffu + ((u >> 16) & 1u);           // round-to-nearest-even
  return (unsigned short)(u >> 16);
}

// A fragment (16x32 bf16, MxK) gathered from an LDS tile stored row-major [16][ldk].
// Per ISA: lanes 0-15: VGPR v<4 -> K = 2v,2v+1 ; v>=4 -> K = 16+2(v-4),...
//          lanes 16-31 add +8 to the K base within each half.
__device__ __forceinline__ v16bf load_a_lds(const unsigned short* lds, int ldk, int k0, int lane) {
  FragB16 f;
  int m   = lane & 15;
  int hi8 = (lane & 16) ? 8 : 0;
  const unsigned short* row = lds + m * ldk + k0 + hi8;
#pragma unroll
  for (int v = 0; v < 4; ++v) f.u[v]     = *(const unsigned int*)(row + 2 * v);
#pragma unroll
  for (int v = 0; v < 4; ++v) f.u[v + 4] = *(const unsigned int*)(row + 16 + 2 * v);
  return f.v;
}

// B fragment (32x16 bf16, KxN) from global weights stored TRANSPOSED [N][ldk] bf16.
// lanes 0-15: v<4 -> K=2v,2v+1; v>=4 -> K=16+2(v-4),...; lanes 16-31 add +8. N = lane&15.
__device__ __forceinline__ v16bf load_b_wt(const unsigned short* wT, int ldk, int n0, int k0, int lane) {
  FragB16 f;
  int n    = n0 + (lane & 15);
  int hi16 = (lane & 16) ? 8 : 0;
  const unsigned short* base = wT + (size_t)n * ldk + k0 + hi16;
#pragma unroll
  for (int v = 0; v < 8; ++v) f.u[v] = *(const unsigned int*)(base + 2 * (v & 3) + ((v >> 2) << 4));
  return f.v;
}

__device__ __forceinline__ v8f wmma_bf16(v16bf a, v16bf b, v8f c) {
  return __builtin_amdgcn_wmma_f32_16x16x32_bf16(false, a, false, b, (short)0, c, false, false);
}

// ---------- weight convert + transpose: f32 [in][out] -> bf16 [out][in] ----------
__global__ void __launch_bounds__(256)
k_cvt_t(const float* __restrict__ src, unsigned short* __restrict__ dst, int in_dim, int out_dim) {
  int idx = blockIdx.x * 256 + threadIdx.x;
  if (idx >= in_dim * out_dim) return;
  int i = idx / out_dim, o = idx % out_dim;
  dst[(size_t)o * in_dim + i] = f2bf(src[idx]);
}

// ---------- input projection: X[8192,128] @ W_in -> xbuf[8192,256] ----------
__global__ void __launch_bounds__(256)
k_in_proj(const float* __restrict__ X, const unsigned short* __restrict__ WT, float* __restrict__ Y) {
  __shared__ unsigned short lsx[16 * kDin];
  int tid = threadIdx.x, row0 = blockIdx.x * 16;
  const float* xs = X + (size_t)row0 * kDin;
#pragma unroll
  for (int e = 0; e < 8; ++e) { int i = tid + e * 256; lsx[i] = f2bf(xs[i]); }
  __syncthreads();
  int wave = tid >> 5, lane = tid & 31;
#pragma unroll
  for (int t = 0; t < 2; ++t) {
    int nt = wave * 2 + t;
    FragF32 acc;
#pragma unroll
    for (int r = 0; r < 8; ++r) acc.f[r] = 0.f;
    for (int k0 = 0; k0 < kDin; k0 += 32)
      acc.v = wmma_bf16(load_a_lds(lsx, kDin, k0, lane), load_b_wt(WT, kDin, nt * 16, k0, lane), acc.v);
    int n = nt * 16 + (lane & 15), mb = (lane & 16) ? 8 : 0;
    float* yb = Y + (size_t)row0 * kD;
#pragma unroll
    for (int r = 0; r < 8; ++r) yb[(size_t)(mb + r) * kD + n] = acc.f[r];
  }
}

// ---------- QKV projection for all tokens ----------
__global__ void __launch_bounds__(256)
k_qkv(const float* __restrict__ X,
      const unsigned short* __restrict__ WqT, const unsigned short* __restrict__ WkT,
      const unsigned short* __restrict__ WvT,
      const float* __restrict__ bq, const float* __restrict__ bk, const float* __restrict__ bv,
      float* __restrict__ Q, float* __restrict__ K, float* __restrict__ V) {
  __shared__ unsigned short lsx[16 * kD];
  int tid = threadIdx.x, row0 = blockIdx.x * 16;
  const float* xs = X + (size_t)row0 * kD;
#pragma unroll
  for (int e = 0; e < 16; ++e) { int i = tid + e * 256; lsx[i] = f2bf(xs[i]); }
  __syncthreads();
  int wave = tid >> 5, lane = tid & 31;
  for (int j = 0; j < 6; ++j) {
    int job = wave * 6 + j;                 // 48 jobs = 3 mats x 16 n-tiles
    int mat = job >> 4, nt = job & 15;
    const unsigned short* wT = (mat == 0) ? WqT : (mat == 1) ? WkT : WvT;
    const float* bias = (mat == 0) ? bq : (mat == 1) ? bk : bv;
    float* out = (mat == 0) ? Q : (mat == 1) ? K : V;
    FragF32 acc;
#pragma unroll
    for (int r = 0; r < 8; ++r) acc.f[r] = 0.f;
    for (int k0 = 0; k0 < kD; k0 += 32)
      acc.v = wmma_bf16(load_a_lds(lsx, kD, k0, lane), load_b_wt(wT, kD, nt * 16, k0, lane), acc.v);
    int n = nt * 16 + (lane & 15), mb = (lane & 16) ? 8 : 0;
    float bb = bias[n];
    float* ob = out + (size_t)row0 * kD;
#pragma unroll
    for (int r = 0; r < 8; ++r) ob[(size_t)(mb + r) * kD + n] = acc.f[r] + bb;
  }
}

// ---------- geometry bias helper ----------
__device__ __forceinline__ float pos_bias(float lon1, float lat1, float sl1, float cl1,
                                          float lon2, float lat2,
                                          const float* __restrict__ Wpos,
                                          const float* __restrict__ Wb, int h) {
  float dlon = lon2 - lon1;
  float sl2 = sinf(lat2), cl2 = cosf(lat2), cdl = cosf(dlon);
  float cosd = fminf(fmaxf(sl1 * sl2 + cl1 * cl2 * cdl, -1.f + 1e-6f), 1.f - 1e-6f);
  float dd  = acosf(cosd);
  float ang = atan2f(sinf(dlon) * cl2, cl1 * sl2 - sl1 * cl2 * cdl);
  float bias = 0.f;
#pragma unroll
  for (int e = 0; e < kPED; ++e) {
    float z = dd * Wpos[e] + ang * Wpos[kPED + e];
    bias += (z / (1.f + expf(-z))) * Wb[e * kH + h];   // silu(z) * Wb[e][h]
  }
  return bias;
}

// ---------- neighborhood attention: thread = (token, head) ----------
__global__ void __launch_bounds__(256)
k_nh_attn(const float* __restrict__ Q, const float* __restrict__ K, const float* __restrict__ V,
          const int* __restrict__ adjc, const float* __restrict__ coords,
          const float* __restrict__ Wpos, const float* __restrict__ Wb,
          float* __restrict__ Aout) {
  int gid = blockIdx.x * 256 + threadIdx.x;
  if (gid >= kNtok * kH) return;
  int tok = gid >> 3, h = gid & 7;
  const float* lon = coords;
  const float* lat = coords + kNtok;
  float lon1 = lon[tok], lat1 = lat[tok];
  float sl1 = sinf(lat1), cl1 = cosf(lat1);
  float q[kHD];
  const float* qp = Q + (size_t)tok * kD + h * kHD;
#pragma unroll
  for (int d = 0; d < kHD; ++d) q[d] = qp[d];
  int nb[kNH]; float sc[kNH];
  for (int j = 0; j < kNH; ++j) {
    int t2 = adjc[tok * kNH + j];
    nb[j] = t2;
    float bias = pos_bias(lon1, lat1, sl1, cl1, lon[t2], lat[t2], Wpos, Wb, h);
    const float* kp = K + (size_t)t2 * kD + h * kHD;
    float dot = 0.f;
#pragma unroll
    for (int d = 0; d < kHD; ++d) dot += q[d] * kp[d];
    sc[j] = dot * 0.1767766952966369f + bias;         // 1/sqrt(32)
  }
  float mx = sc[0];
#pragma unroll
  for (int j = 1; j < kNH; ++j) mx = fmaxf(mx, sc[j]);
  float ssum = 0.f;
#pragma unroll
  for (int j = 0; j < kNH; ++j) { sc[j] = expf(sc[j] - mx); ssum += sc[j]; }
  float inv = 1.f / ssum;
  float acc[kHD];
#pragma unroll
  for (int d = 0; d < kHD; ++d) acc[d] = 0.f;
  for (int j = 0; j < kNH; ++j) {
    float w = sc[j] * inv;
    const float* vp = V + (size_t)nb[j] * kD + h * kHD;
#pragma unroll
    for (int d = 0; d < kHD; ++d) acc[d] += w * vp[d];
  }
  float* op = Aout + (size_t)tok * kD + h * kHD;
#pragma unroll
  for (int d = 0; d < kHD; ++d) op[d] = acc[d];
}

// ---------- sequence attention: block = 16-token chunk, thread = (head, q-row) ----------
__global__ void __launch_bounds__(128)
k_seq_attn(const float* __restrict__ Q, const float* __restrict__ K, const float* __restrict__ V,
           const float* __restrict__ coords, const float* __restrict__ Wpos,
           const float* __restrict__ Wb, float* __restrict__ Aout) {
  int tid = threadIdx.x;                // 128 = 8 heads x 16 rows
  int h = tid >> 4, qr = tid & 15;
  int base = blockIdx.x * kS;
  int tq = base + qr;
  const float* lon = coords;
  const float* lat = coords + kNtok;
  float lon1 = lon[tq], lat1 = lat[tq];
  float sl1 = sinf(lat1), cl1 = cosf(lat1);
  float q[kHD];
  const float* qp = Q + (size_t)tq * kD + h * kHD;
#pragma unroll
  for (int d = 0; d < kHD; ++d) q[d] = qp[d];
  float sc[kS];
  for (int kc = 0; kc < kS; ++kc) {
    int tk = base + kc;
    float bias = pos_bias(lon1, lat1, sl1, cl1, lon[tk], lat[tk], Wpos, Wb, h);
    const float* kp = K + (size_t)tk * kD + h * kHD;
    float dot = 0.f;
#pragma unroll
    for (int d = 0; d < kHD; ++d) dot += q[d] * kp[d];
    sc[kc] = dot * 0.1767766952966369f + bias;
  }
  float mx = sc[0];
#pragma unroll
  for (int kc = 1; kc < kS; ++kc) mx = fmaxf(mx, sc[kc]);
  float ssum = 0.f;
#pragma unroll
  for (int kc = 0; kc < kS; ++kc) { sc[kc] = expf(sc[kc] - mx); ssum += sc[kc]; }
  float inv = 1.f / ssum;
  float acc[kHD];
#pragma unroll
  for (int d = 0; d < kHD; ++d) acc[d] = 0.f;
  for (int kc = 0; kc < kS; ++kc) {
    float w = sc[kc] * inv;
    const float* vp = V + (size_t)(base + kc) * kD + h * kHD;
#pragma unroll
    for (int d = 0; d < kHD; ++d) acc[d] += w * vp[d];
  }
  float* op = Aout + (size_t)tq * kD + h * kHD;
#pragma unroll
  for (int d = 0; d < kHD; ++d) op[d] = acc[d];
}

// ---------- per-row layernorm stats over 256 cols (16 threads per row) ----------
__device__ __forceinline__ void row_stats(const float* ls, int tid, float* redS, float* redQ, float* mv) {
  int m = tid >> 4, j = tid & 15;
  float s = 0.f, s2 = 0.f;
#pragma unroll
  for (int e = 0; e < 16; ++e) {
    float x = ls[m * kD + (e << 4) + j];
    s += x; s2 += x * x;
  }
  redS[m * 16 + j] = s;
  redQ[m * 16 + j] = s2;
  __syncthreads();
  if (j == 0) {
    float S = 0.f, S2 = 0.f;
#pragma unroll
    for (int e = 0; e < 16; ++e) { S += redS[m * 16 + e]; S2 += redQ[m * 16 + e]; }
    float mean = S * (1.f / kD);
    float var  = S2 * (1.f / kD) - mean * mean;
    mv[m * 2]     = mean;
    mv[m * 2 + 1] = rsqrtf(var + 1e-5f);
  }
  __syncthreads();
}

// ---------- block tail: O-proj + LN1 + MLP + LN2, fused per 16-token tile ----------
__global__ void __launch_bounds__(256)
k_block_tail(const float* __restrict__ Xin, const float* __restrict__ Attn,
             const unsigned short* __restrict__ WoT, const float* __restrict__ bo,
             const float* __restrict__ g1, const float* __restrict__ b1,
             const unsigned short* __restrict__ W1T, const unsigned short* __restrict__ W2T,
             const float* __restrict__ g2, const float* __restrict__ b2,
             float* __restrict__ Xout) {
  __shared__ unsigned short ls_a[16 * kD];    // attn tile, bf16
  __shared__ float          ls_r[16 * kD];    // x residual -> h(f32) -> h+mlp
  __shared__ unsigned short ls_h[16 * kD];    // h, bf16 (GEMM1 A)
  __shared__ unsigned short ls_f[16 * kFF];   // silu(h@W1), bf16 (GEMM2 A)
  __shared__ float redS[256], redQ[256], mv[32];

  int tid = threadIdx.x, row0 = blockIdx.x * 16;
  int wave = tid >> 5, lane = tid & 31;
  int m16 = tid >> 4, j16 = tid & 15;

  { // stage attn (bf16) + x (f32)
    const float* as = Attn + (size_t)row0 * kD;
    const float* xs = Xin + (size_t)row0 * kD;
#pragma unroll
    for (int e = 0; e < 16; ++e) {
      int i = tid + e * 256;
      ls_a[i] = f2bf(as[i]);
      ls_r[i] = xs[i];
    }
  }
  __syncthreads();

  // O-projection: ls_r += attn @ Wo + bo
#pragma unroll
  for (int t = 0; t < 2; ++t) {
    int nt = wave * 2 + t;
    FragF32 acc;
#pragma unroll
    for (int r = 0; r < 8; ++r) acc.f[r] = 0.f;
    for (int k0 = 0; k0 < kD; k0 += 32)
      acc.v = wmma_bf16(load_a_lds(ls_a, kD, k0, lane), load_b_wt(WoT, kD, nt * 16, k0, lane), acc.v);
    int n = nt * 16 + (lane & 15), mb = (lane & 16) ? 8 : 0;
    float bb = bo[n];
#pragma unroll
    for (int r = 0; r < 8; ++r) ls_r[(mb + r) * kD + n] += acc.f[r] + bb;
  }
  __syncthreads();

  // LN1 -> ls_r (f32 h) + ls_h (bf16)
  row_stats(ls_r, tid, redS, redQ, mv);
  {
    float mean = mv[m16 * 2], rstd = mv[m16 * 2 + 1];
#pragma unroll
    for (int e = 0; e < 16; ++e) {
      int n = (e << 4) + j16, idx = m16 * kD + n;
      float hh = (ls_r[idx] - mean) * rstd * g1[n] + b1[n];
      ls_r[idx] = hh;
      ls_h[idx] = f2bf(hh);
    }
  }
  __syncthreads();

  // GEMM1: silu(h @ W1) -> ls_f
#pragma unroll
  for (int t = 0; t < 8; ++t) {
    int nt = wave * 8 + t;                  // 64 n-tiles
    FragF32 acc;
#pragma unroll
    for (int r = 0; r < 8; ++r) acc.f[r] = 0.f;
    for (int k0 = 0; k0 < kD; k0 += 32)
      acc.v = wmma_bf16(load_a_lds(ls_h, kD, k0, lane), load_b_wt(W1T, kD, nt * 16, k0, lane), acc.v);
    int n = nt * 16 + (lane & 15), mb = (lane & 16) ? 8 : 0;
#pragma unroll
    for (int r = 0; r < 8; ++r) {
      float z = acc.f[r];
      ls_f[(mb + r) * kFF + n] = f2bf(z / (1.f + expf(-z)));
    }
  }
  __syncthreads();

  // GEMM2: ls_r += f @ W2
#pragma unroll
  for (int t = 0; t < 2; ++t) {
    int nt = wave * 2 + t;
    FragF32 acc;
#pragma unroll
    for (int r = 0; r < 8; ++r) acc.f[r] = 0.f;
    for (int k0 = 0; k0 < kFF; k0 += 32)
      acc.v = wmma_bf16(load_a_lds(ls_f, kFF, k0, lane), load_b_wt(W2T, kFF, nt * 16, k0, lane), acc.v);
    int n = nt * 16 + (lane & 15), mb = (lane & 16) ? 8 : 0;
#pragma unroll
    for (int r = 0; r < 8; ++r) ls_r[(mb + r) * kD + n] += acc.f[r];
  }
  __syncthreads();

  // LN2 -> Xout
  row_stats(ls_r, tid, redS, redQ, mv);
  {
    float mean = mv[m16 * 2], rstd = mv[m16 * 2 + 1];
    float* xo = Xout + (size_t)row0 * kD;
#pragma unroll
    for (int e = 0; e < 16; ++e) {
      int n = (e << 4) + j16, idx = m16 * kD + n;
      xo[idx] = (ls_r[idx] - mean) * rstd * g2[n] + b2[n];
    }
  }
}

extern "C" void kernel_launch(void* const* d_in, const int* in_sizes, int n_in,
                              void* d_out, int out_size, void* d_ws, size_t ws_size,
                              hipStream_t stream) {
  (void)in_sizes; (void)n_in; (void)out_size; (void)ws_size;
  // setup_inputs() order: x, indices_global_level, adjc, sample, coordinates, W_in, W_pos,
  //                       nh_params (15 leaves), seq_params (15 leaves)
  // leaf insertion order: Wq bq Wk bk Wv bv Wo bo Wb W1 W2 g1 b1 g2 b2
  const float* x_in   = (const float*)d_in[0];
  const int*   adjc   = (const int*)  d_in[2];   // global token indices (sample offset is identity here)
  const float* coords = (const float*)d_in[4];
  const float* W_in   = (const float*)d_in[5];
  const float* W_pos  = (const float*)d_in[6];
  const int baseIdx[2] = {7, 22};                // nh, seq

  char* ws = (char*)d_ws;
  size_t off = 0;
  auto carve = [&](size_t bytes) { char* p = ws + off; off += (bytes + 255) & ~(size_t)255; return p; };
  float* xbuf = (float*)carve((size_t)kNtok * kD * 4);
  float* qb   = (float*)carve((size_t)kNtok * kD * 4);
  float* kb   = (float*)carve((size_t)kNtok * kD * 4);
  float* vb   = (float*)carve((size_t)kNtok * kD * 4);
  float* attn = (float*)carve((size_t)kNtok * kD * 4);
  unsigned short* WinT = (unsigned short*)carve((size_t)kDin * kD * 2);
  unsigned short *WqT[2][2], *WkT[2][2], *WvT[2][2], *WoT[2][2], *W1T[2][2], *W2T[2][2];
  for (int g = 0; g < 2; ++g)
    for (int l = 0; l < kL; ++l) {
      WqT[g][l] = (unsigned short*)carve((size_t)kD * kD * 2);
      WkT[g][l] = (unsigned short*)carve((size_t)kD * kD * 2);
      WvT[g][l] = (unsigned short*)carve((size_t)kD * kD * 2);
      WoT[g][l] = (unsigned short*)carve((size_t)kD * kD * 2);
      W1T[g][l] = (unsigned short*)carve((size_t)kD * kFF * 2);
      W2T[g][l] = (unsigned short*)carve((size_t)kFF * kD * 2);
    }

  auto cvt = [&](const float* src, unsigned short* dst, int in_dim, int out_dim) {
    int n = in_dim * out_dim;
    k_cvt_t<<<dim3((n + 255) / 256), dim3(256), 0, stream>>>(src, dst, in_dim, out_dim);
  };
  cvt(W_in, WinT, kDin, kD);
  for (int g = 0; g < 2; ++g) {
    int b = baseIdx[g];
    for (int l = 0; l < kL; ++l) {
      cvt((const float*)d_in[b + 0]  + (size_t)l * kD * kD,  WqT[g][l], kD, kD);
      cvt((const float*)d_in[b + 2]  + (size_t)l * kD * kD,  WkT[g][l], kD, kD);
      cvt((const float*)d_in[b + 4]  + (size_t)l * kD * kD,  WvT[g][l], kD, kD);
      cvt((const float*)d_in[b + 6]  + (size_t)l * kD * kD,  WoT[g][l], kD, kD);
      cvt((const float*)d_in[b + 9]  + (size_t)l * kD * kFF, W1T[g][l], kD, kFF);
      cvt((const float*)d_in[b + 10] + (size_t)l * kFF * kD, W2T[g][l], kFF, kD);
    }
  }

  k_in_proj<<<dim3(kNtok / 16), dim3(256), 0, stream>>>(x_in, WinT, xbuf);

  for (int l = 0; l < kL; ++l) {
    for (int g = 0; g < 2; ++g) {
      int b = baseIdx[g];
      const float* bq = (const float*)d_in[b + 1]  + (size_t)l * kD;
      const float* bk = (const float*)d_in[b + 3]  + (size_t)l * kD;
      const float* bv = (const float*)d_in[b + 5]  + (size_t)l * kD;
      const float* bo = (const float*)d_in[b + 7]  + (size_t)l * kD;
      const float* Wb = (const float*)d_in[b + 8]  + (size_t)l * kPED * kH;
      const float* g1 = (const float*)d_in[b + 11] + (size_t)l * kD;
      const float* b1 = (const float*)d_in[b + 12] + (size_t)l * kD;
      const float* g2 = (const float*)d_in[b + 13] + (size_t)l * kD;
      const float* b2 = (const float*)d_in[b + 14] + (size_t)l * kD;

      k_qkv<<<dim3(kNtok / 16), dim3(256), 0, stream>>>(xbuf, WqT[g][l], WkT[g][l], WvT[g][l],
                                                        bq, bk, bv, qb, kb, vb);
      if (g == 0)
        k_nh_attn<<<dim3(kNtok * kH / 256), dim3(256), 0, stream>>>(qb, kb, vb, adjc, coords,
                                                                    W_pos, Wb, attn);
      else
        k_seq_attn<<<dim3(kNtok / kS), dim3(128), 0, stream>>>(qb, kb, vb, coords, W_pos, Wb, attn);

      float* xout = (l == kL - 1 && g == 1) ? (float*)d_out : xbuf;
      k_block_tail<<<dim3(kNtok / 16), dim3(256), 0, stream>>>(xbuf, attn, WoT[g][l], bo, g1, b1,
                                                               W1T[g][l], W2T[g][l], g2, b2, xout);
    }
  }
}
